// MRMBlock_56143812493388
// MI455X (gfx1250) — matrix operand order repaired and checked by reading, not compiled, not run on hardware
//
#include <hip/hip_runtime.h>
#include <hip/hip_bf16.h>
#include <stdint.h>

// ---------------------------------------------------------------------------
// MI455X (gfx1250) Mamba block + RMSNorm + SwiGLU MLP.
// GEMMs: bf16 WMMA (v_wmma_f32_16x16x32_bf16), f32 accumulation.
//   - A (activation) tiles: per-lane async global->LDS (ASYNCcnt)
//   - B (weight) tiles: Tensor Data Mover tensor_load_to_lds (TENSORcnt),
//     issued by ONE wave per workgroup (scalar-uniform guard),
//     descriptor-padded to an 80B LDS row stride (bank-conflict free).
// Scan / norms / activations stay fp32 VALU.
// ---------------------------------------------------------------------------

typedef __bf16 bf16;
typedef __bf16 v16bf __attribute__((ext_vector_type(16)));
typedef float  v8f   __attribute__((ext_vector_type(8)));
typedef unsigned int v4u __attribute__((ext_vector_type(4)));
typedef unsigned int v8u __attribute__((ext_vector_type(8)));

union FragBF { v16bf v; v4u q[2]; };

// ---- CDNA5 async global->LDS copy (16 bytes / lane), tracked by ASYNCcnt ---
__device__ __forceinline__ void async_cp_b128(unsigned lds_byte_addr,
                                              const void* sbase,
                                              unsigned voffset_bytes) {
  asm volatile("global_load_async_to_lds_b128 %0, %1, %2"
               :: "v"(lds_byte_addr), "v"(voffset_bytes), "s"(sbase)
               : "memory");
}

template <int N>
__device__ __forceinline__ void wait_asynccnt() {
  asm volatile("s_wait_asynccnt %0" :: "n"(N) : "memory");
}

// ---- CDNA5 Tensor Data Mover: 2D tile global->LDS, tracked by TENSORcnt ---
// D# group0: {count=1, lds_addr, global_addr[31:0], global_addr[56:32]|type=2}
// D# group1: data_size=2B, pad_enable, pad_interval=16dw, pad_amount=4dw,
//            tensor_dim0=K, tensor_dim1=N, tile_dim0=32, tile_dim1=128,
//            tensor_dim0_stride=K.
__device__ __forceinline__ void tdm_load_tile(unsigned lds_byte_addr,
                                              const void* gptr,
                                              unsigned K, unsigned Nrows) {
  unsigned long long ga = (unsigned long long)gptr;
  v4u g0;
  g0[0] = 1u;                                   // count=1, user descriptor
  g0[1] = lds_byte_addr;                        // lds_addr
  g0[2] = (unsigned)ga;                         // global_addr[31:0]
  g0[3] = (unsigned)((ga >> 32) & 0x01FFFFFFu)  // global_addr[56:32]
          | 0x80000000u;                        // type = 2 ("image")
  v8u g1;
  g1[0] = (1u << 16)        // data_size = 1 -> 2 bytes (bf16)
        | (1u << 20)        // pad_enable
        | (3u << 22)        // pad_interval code 3 -> 16 DWORDs (64B) of data
        | (3u << 25);       // pad_amount  code 3 -> 4 DWORDs (16B) of pad
  g1[1] = (K & 0xFFFFu) << 16;                  // tensor_dim0[15:0]
  g1[2] = (K >> 16) | ((Nrows & 0xFFFFu) << 16);// tensor_dim0[31:16], dim1 lo
  g1[3] = (Nrows >> 16) | (32u << 16);          // dim1 hi, tile_dim0 = 32
  g1[4] = 128u;                                 // tile_dim1 = 128 rows
  g1[5] = K;                                    // tensor_dim0_stride[31:0]
  g1[6] = 0u;                                   // stride hi, dim1_stride lo
  g1[7] = 0u;
  asm volatile("tensor_load_to_lds %0, %1" :: "s"(g0), "s"(g1) : "memory");
}

// ---------------------------------------------------------------------------
// GEMM: C[M,N] (f32) = A[M,K](bf16, row stride lda) * W[N,K](bf16)^T
// Requires M%256==0, N%128==0, K%32==0 (all call sites padded accordingly).
// Block = 256 threads (8 waves, 4x2), tile 256x128, wave tile 64x64,
// K-step 32, double buffered (A: async-to-LDS, B: TDM by leader wave).
// ---------------------------------------------------------------------------
#define BM 256
#define BN 128
#define BK 32
#define LDSROW 40   // 32 data bf16 + 8 pad -> 80B row

__global__ __launch_bounds__(256)
void gemm_bf16_nt(const bf16* __restrict__ A, const bf16* __restrict__ W,
                  float* __restrict__ C, int M, int N, int K, int lda) {
  __shared__ __align__(16) bf16 sA[2][BM * LDSROW];
  __shared__ __align__(16) bf16 sB[2][BN * LDSROW];

  const int tid   = threadIdx.x;
  const int lane  = tid & 31;
  const int lrow  = lane & 15;
  const int lhalf = lane >> 4;
  const int wave  = tid >> 5;
  const int wm    = (wave & 3) * 64;   // wave M offset inside tile
  const int wn    = (wave >> 2) * 64;  // wave N offset inside tile

  // Wave-uniform leader predicate (scalar branch, no EXEC masking): only the
  // leader wave drives the Tensor Data Mover (TDM ignores EXEC, so a lane-
  // predicated guard would issue 8 duplicate DMAs per stage).
  const bool leader = (__builtin_amdgcn_readfirstlane(tid) < 32);

  const int m0 = blockIdx.y * BM;
  const int n0 = blockIdx.x * BN;

  const unsigned ldsA = (unsigned)(unsigned long long)(&sA[0][0]);
  const unsigned ldsB = (unsigned)(unsigned long long)(&sB[0][0]);
  const unsigned ASTAGE = BM * LDSROW * 2;   // bytes per A stage buffer
  const unsigned BSTAGE = BN * LDSROW * 2;   // bytes per B stage buffer

  // A tile: 256 rows x 32 bf16 = 1024 16B-chunks; 4 async copies per thread.
  auto issueA = [&](int kt, int buf) {
#pragma unroll
    for (int i = 0; i < 4; ++i) {
      const int c   = tid + i * 256;
      const int row = c >> 2;
      const int kc  = (c & 3) * 8;
      const unsigned l = ldsA + (unsigned)buf * ASTAGE +
                         (unsigned)((row * LDSROW + kc) * 2);
      const unsigned ga =
          (unsigned)(((size_t)(m0 + row) * (size_t)lda + (size_t)(kt + kc)) * 2u);
      async_cp_b128(l, A, ga);
    }
  };
  // B tile: one TDM descriptor moves 128 rows x 32 bf16 with LDS padding.
  auto issueB = [&](int kt, int buf) {
    const bf16* g = W + (size_t)n0 * (size_t)K + (size_t)kt;
    tdm_load_tile(ldsB + (unsigned)buf * BSTAGE, g, (unsigned)K, (unsigned)N);
  };

  v8f acc[4][4] = {};

  const int nk = K / BK;
  issueA(0, 0);
  if (leader) issueB(0, 0);
  for (int kt = 0; kt < nk; ++kt) {
    const int buf = kt & 1;
    if (kt + 1 < nk) {
      issueA((kt + 1) * BK, buf ^ 1);
      if (leader) issueB((kt + 1) * BK, buf ^ 1);
      wait_asynccnt<4>();                              // stage kt's A done
      if (leader) __builtin_amdgcn_s_wait_tensorcnt((short)1);     // B done
    } else {
      wait_asynccnt<0>();
      if (leader) __builtin_amdgcn_s_wait_tensorcnt((short)0);
    }
    __syncthreads();        // publish async/TDM LDS fills to all waves

    FragBF af[4], bfg[4];
#pragma unroll
    for (int fm = 0; fm < 4; ++fm) {
      const bf16* p = &sA[buf][(wm + fm * 16 + lrow) * LDSROW];
      af[fm].q[0] = *reinterpret_cast<const v4u*>(p + lhalf * 8);
      af[fm].q[1] = *reinterpret_cast<const v4u*>(p + 16 + lhalf * 8);
    }
#pragma unroll
    for (int fn = 0; fn < 4; ++fn) {
      const bf16* p = &sB[buf][(wn + fn * 16 + lrow) * LDSROW + lhalf * 8];
      bfg[fn].q[0] = *reinterpret_cast<const v4u*>(p);
      bfg[fn].q[1] = *reinterpret_cast<const v4u*>(p + 16);
    }
#pragma unroll
    for (int fm = 0; fm < 4; ++fm)
#pragma unroll
      for (int fn = 0; fn < 4; ++fn)
        acc[fm][fn] = __builtin_amdgcn_wmma_f32_16x16x32_bf16(
            false, af[fm].v, false, bfg[fn].v, (short)0, acc[fm][fn],
            false, false);
    __syncthreads();        // reads of buf done before it is refilled
  }

  // Epilogue: C layout per ISA (VGPR j -> M = 8*lhalf + j, N = lrow)
#pragma unroll
  for (int fm = 0; fm < 4; ++fm)
#pragma unroll
    for (int fn = 0; fn < 4; ++fn) {
      const int col = n0 + wn + fn * 16 + lrow;
#pragma unroll
      for (int j = 0; j < 8; ++j) {
        const int row = m0 + wm + fm * 16 + lhalf * 8 + j;
        C[(size_t)row * (size_t)N + col] = acc[fm][fn][j];
      }
    }
}

// ---------------------------------------------------------------------------
// Elementwise / pointwise kernels (fp32 VALU)
// ---------------------------------------------------------------------------
__device__ __forceinline__ float silu_f(float x) {
  return x / (1.f + __expf(-x));
}

__global__ __launch_bounds__(256)
void k_cvt(const float* __restrict__ in, bf16* __restrict__ out, size_t n) {
  size_t i = (size_t)blockIdx.x * 256 + threadIdx.x;
  if (i < n) out[i] = (bf16)in[i];
}

// x_proj_w (96,2048) -> zero-padded (128,2048) bf16
__global__ __launch_bounds__(256)
void k_cvt_pad(const float* __restrict__ in, bf16* __restrict__ out,
               int rows_in, int cols) {
  size_t i = (size_t)blockIdx.x * 256 + threadIdx.x;
  int r = (int)(i / (size_t)cols);
  int c = (int)(i % (size_t)cols);
  out[i] = (r < rows_in) ? (bf16)in[(size_t)r * cols + c] : (bf16)0.f;
}

// Depthwise causal conv (DC=4) over L on x = xz[:, :, :2048], then SiLU.
__global__ __launch_bounds__(256)
void k_conv_silu(const float* __restrict__ xz, const float* __restrict__ cw,
                 const float* __restrict__ cb, bf16* __restrict__ xcB) {
  size_t idx = (size_t)blockIdx.x * 256 + threadIdx.x;  // (b,l,d) flattened
  int d = (int)(idx & 2047);
  size_t tok = idx >> 11;                               // b*L + l
  int l = (int)(tok & 2047);
  float acc = cb[d];
#pragma unroll
  for (int j = 0; j < 4; ++j) {
    int ll = l + j - 3;
    if (ll >= 0)
      acc += cw[d * 4 + j] * xz[(tok + (size_t)(j - 3)) * 4096 + d];
  }
  xcB[idx] = (bf16)silu_f(acc);
}

// delta = softplus(delta + bias[d])   (in place)
__global__ __launch_bounds__(256)
void k_bias_softplus(float* __restrict__ dl, const float* __restrict__ bias) {
  size_t idx = (size_t)blockIdx.x * 256 + threadIdx.x;
  int d = (int)(idx & 2047);
  float x = dl[idx] + bias[d];
  dl[idx] = (x > 20.f) ? x : log1pf(__expf(x));
}

// Selective scan: thread per (b,d) channel, N=16 state in registers,
// fused +x*D and *silu(z) epilogue; emits bf16 for the out_proj GEMM.
__global__ __launch_bounds__(256)
void k_scan(const float* __restrict__ delta, const bf16* __restrict__ xcB,
            const float* __restrict__ xdbl, const float* __restrict__ xz,
            const float* __restrict__ A_log, const float* __restrict__ Dp,
            bf16* __restrict__ yB) {
  const int t = blockIdx.x * 256 + threadIdx.x;  // 0..8191 = b*2048 + d
  const int b = t >> 11;
  const int d = t & 2047;
  float A[16], st[16];
#pragma unroll
  for (int n = 0; n < 16; ++n) {
    A[n] = -__expf(A_log[d * 16 + n]);
    st[n] = 0.f;
  }
  const float Dv = Dp[d];
  for (int l = 0; l < 2048; ++l) {
    const size_t tok = (size_t)b * 2048 + l;
    const float dt = delta[tok * 2048 + d];
    const float u  = (float)xcB[tok * 2048 + d];
    const float du = dt * u;
    const float* xd = xdbl + tok * 128;   // [0:64)=dt, [64:80)=B, [80:96)=C
    float y = 0.f;
#pragma unroll
    for (int n = 0; n < 16; ++n) {
      st[n] = st[n] * __expf(dt * A[n]) + du * xd[64 + n];
      y += st[n] * xd[80 + n];
    }
    const float z = xz[tok * 4096 + 2048 + d];
    yB[tok * 2048 + d] = (bf16)((y + u * Dv) * silu_f(z));
  }
}

// act = silu(g) * u from gu[M,8192] -> bf16 [M,4096]
__global__ __launch_bounds__(256)
void k_silu_mul(const float* __restrict__ gu, bf16* __restrict__ actB) {
  size_t idx = (size_t)blockIdx.x * 256 + threadIdx.x;
  size_t m = idx >> 12;
  int i = (int)(idx & 4095);
  float g = gu[m * 8192 + i];
  float u = gu[m * 8192 + 4096 + i];
  actB[idx] = (bf16)(silu_f(g) * u);
}

// out = rms_norm(a + b), one block per row of 1024, optional bf16 copy
__global__ __launch_bounds__(256)
void k_add_rmsnorm(const float* __restrict__ a, const float* __restrict__ b,
                   float* __restrict__ out, bf16* __restrict__ outB) {
  __shared__ float red[256];
  const size_t base = (size_t)blockIdx.x * 1024;
  float v[4];
  float ss = 0.f;
#pragma unroll
  for (int i = 0; i < 4; ++i) {
    int c = threadIdx.x + i * 256;
    float x = a[base + c] + b[base + c];
    v[i] = x;
    ss += x * x;
  }
  red[threadIdx.x] = ss;
  __syncthreads();
  for (int s = 128; s > 0; s >>= 1) {
    if ((int)threadIdx.x < s) red[threadIdx.x] += red[threadIdx.x + s];
    __syncthreads();
  }
  const float scale = rsqrtf(red[0] * (1.f / 1024.f) + 1e-5f);
#pragma unroll
  for (int i = 0; i < 4; ++i) {
    int c = threadIdx.x + i * 256;
    float y = v[i] * scale;
    out[base + c] = y;
    if (outB) outB[base + c] = (bf16)y;
  }
}

// ---------------------------------------------------------------------------
// Host orchestration
// ---------------------------------------------------------------------------
extern "C" void kernel_launch(void* const* d_in, const int* in_sizes, int n_in,
                              void* d_out, int out_size, void* d_ws, size_t ws_size,
                              hipStream_t stream) {
  (void)in_sizes; (void)n_in; (void)out_size; (void)ws_size;
  const size_t M = 8192;  // B*L

  const float* hs      = (const float*)d_in[1];   // (4,2048,1024)
  const float* inW     = (const float*)d_in[2];   // (4096,1024)
  const float* convW   = (const float*)d_in[3];   // (2048,1,4)
  const float* convBia = (const float*)d_in[4];   // (2048)
  const float* xpW     = (const float*)d_in[5];   // (96,2048)
  const float* dtW     = (const float*)d_in[6];   // (2048,64)
  const float* dtB     = (const float*)d_in[7];   // (2048)
  const float* Alog    = (const float*)d_in[8];   // (2048,16)
  const float* Dpar    = (const float*)d_in[9];   // (2048)
  const float* outW    = (const float*)d_in[10];  // (1024,2048)
  const float* guW     = (const float*)d_in[11];  // (8192,1024)
  const float* dnW     = (const float*)d_in[12];  // (1024,4096)
  float* outp          = (float*)d_out;           // (4,2048,1024)

  char* p = (char*)d_ws;
  auto take = [&](size_t bytes) -> char* {
    char* r = p;
    p += (bytes + 255) & ~(size_t)255;
    return r;
  };
  bf16* wInB  = (bf16*)take(4096ull * 1024 * 2);
  bf16* wXpB  = (bf16*)take(128ull * 2048 * 2);   // padded 96 -> 128 rows
  bf16* wDtB  = (bf16*)take(2048ull * 64 * 2);
  bf16* wOutB = (bf16*)take(1024ull * 2048 * 2);
  bf16* wGuB  = (bf16*)take(8192ull * 1024 * 2);
  bf16* wDnB  = (bf16*)take(1024ull * 4096 * 2);
  bf16* hsB   = (bf16*)take(M * 1024 * 2);
  float* xz   = (float*)take(M * 4096 * 4);
  bf16* xcB   = (bf16*)take(M * 2048 * 2);
  float* xdbl = (float*)take(M * 128 * 4);
  bf16* xdblB = (bf16*)take(M * 128 * 2);
  float* delta= (float*)take(M * 2048 * 4);
  bf16* yB    = (bf16*)take(M * 2048 * 2);
  float* mres = (float*)take(M * 1024 * 4);
  float* h1   = (float*)take(M * 1024 * 4);
  bf16* h1B   = (bf16*)take(M * 1024 * 2);
  float* gu   = (float*)take(M * 8192 * 4);
  bf16* actB  = (bf16*)take(M * 4096 * 2);
  float* mlp  = (float*)take(M * 1024 * 4);

  // --- bf16 conversions of activations & weights -------------------------
  k_cvt<<<32768, 256, 0, stream>>>(hs, hsB, M * 1024);
  k_cvt<<<16384, 256, 0, stream>>>(inW, wInB, 4096ull * 1024);
  k_cvt_pad<<<1024, 256, 0, stream>>>(xpW, wXpB, 96, 2048);
  k_cvt<<<512, 256, 0, stream>>>(dtW, wDtB, 2048ull * 64);
  k_cvt<<<8192, 256, 0, stream>>>(outW, wOutB, 1024ull * 2048);
  k_cvt<<<32768, 256, 0, stream>>>(guW, wGuB, 8192ull * 1024);
  k_cvt<<<16384, 256, 0, stream>>>(dnW, wDnB, 1024ull * 4096);

  // --- Mamba -------------------------------------------------------------
  gemm_bf16_nt<<<dim3(32, 32), 256, 0, stream>>>(hsB, wInB, xz,
                                                 (int)M, 4096, 1024, 1024);
  k_conv_silu<<<65536, 256, 0, stream>>>(xz, convW, convBia, xcB);
  gemm_bf16_nt<<<dim3(1, 32), 256, 0, stream>>>(xcB, wXpB, xdbl,
                                                (int)M, 128, 2048, 2048);
  k_cvt<<<4096, 256, 0, stream>>>(xdbl, xdblB, M * 128);
  gemm_bf16_nt<<<dim3(16, 32), 256, 0, stream>>>(xdblB, wDtB, delta,
                                                 (int)M, 2048, 64, 128);
  k_bias_softplus<<<65536, 256, 0, stream>>>(delta, dtB);
  k_scan<<<32, 256, 0, stream>>>(delta, xcB, xdbl, xz, Alog, Dpar, yB);
  gemm_bf16_nt<<<dim3(8, 32), 256, 0, stream>>>(yB, wOutB, mres,
                                                (int)M, 1024, 2048, 2048);

  // --- residual + RMSNorm ------------------------------------------------
  k_add_rmsnorm<<<8192, 256, 0, stream>>>(hs, mres, h1, h1B);

  // --- SwiGLU MLP ---------------------------------------------------------
  gemm_bf16_nt<<<dim3(64, 32), 256, 0, stream>>>(h1B, wGuB, gu,
                                                 (int)M, 8192, 1024, 1024);
  k_silu_mul<<<131072, 256, 0, stream>>>(gu, actB);
  gemm_bf16_nt<<<dim3(8, 32), 256, 0, stream>>>(actB, wDnB, mlp,
                                                (int)M, 1024, 4096, 4096);

  // --- final residual + RMSNorm -> d_out ----------------------------------
  k_add_rmsnorm<<<8192, 256, 0, stream>>>(h1, mlp, outp, nullptr);
}
